// TransformerModel_43138651521029
// MI455X (gfx1250) — matrix operand order repaired and checked by reading, not compiled, not run on hardware
//
#include <hip/hip_runtime.h>
#include <math.h>

// ---------------------------------------------------------------------------
// Problem constants (match reference)
// ---------------------------------------------------------------------------
#define SEQL    2048
#define NBATCH  8
#define DMODEL  256
#define NHEADS  8
#define DHEAD   32
#define NHID    1024
#define NLAYERS 4
#define LOUT    50
#define MROWS   (NBATCH * SEQL)   // 16384 token rows

typedef __attribute__((ext_vector_type(16))) _Float16 v16h;
typedef __attribute__((ext_vector_type(8)))  _Float16 v8h;
typedef __attribute__((ext_vector_type(8)))  float    v8f;

static __device__ __forceinline__ v8f wmma_f16(v16h a, v16h b, v8f c) {
    // D = A(16x32 f16) * B(32x16 f16) + C(16x16 f32)
    return __builtin_amdgcn_wmma_f32_16x16x32_f16(false, a, false, b,
                                                  (short)0, c, false, false);
}

// CDNA5 async Global->LDS copy (16B per lane), tracked by ASYNCcnt.
// lds: workgroup-relative LDS byte address (low 32 bits of flat shared ptr).
static __device__ __forceinline__ void async_copy16(unsigned lds,
                                                    const void* gaddr) {
    asm volatile("global_load_async_to_lds_b128 %0, %1, off"
                 :: "v"(lds), "v"((unsigned long long)(size_t)gaddr)
                 : "memory");
}

static __device__ __forceinline__ void wait_async0() {
#if __has_builtin(__builtin_amdgcn_s_wait_asynccnt)
    __builtin_amdgcn_s_wait_asynccnt(0);
#else
    asm volatile("s_wait_asynccnt 0x0" ::: "memory");
#endif
}

static __device__ __forceinline__ unsigned lds_addr(const void* p) {
    return (unsigned)(size_t)p;   // low 32 bits of flat shared ptr = LDS offset
}

// ---------------------------------------------------------------------------
// fp32 -> f16 weight conversion
// ---------------------------------------------------------------------------
__global__ void cvt_f16_kernel(const float* __restrict__ src,
                               _Float16* __restrict__ dst, int n) {
    int i = blockIdx.x * 256 + threadIdx.x;
    if (i < n) dst[i] = (_Float16)src[i];
}

// ---------------------------------------------------------------------------
// Embedding: x[row,d] = inputs[row,0] * W_emb[d] + b_emb[d]
// ---------------------------------------------------------------------------
__global__ void embed_kernel(const float* __restrict__ inp,
                             const float* __restrict__ W_emb,
                             const float* __restrict__ b_emb,
                             float* __restrict__ xf,
                             _Float16* __restrict__ xh) {
    int i   = blockIdx.x * 256 + threadIdx.x;   // i < MROWS*DMODEL
    int row = i >> 8;
    int d   = i & 255;
    float v = inp[(size_t)row * 4];             // channel 0
    float o = v * W_emb[d] + b_emb[d];
    xf[i] = o;
    xh[i] = (_Float16)o;
}

// ---------------------------------------------------------------------------
// WMMA GEMM:  out[M,N] = A[M,K](f16) @ W[N,K]^T(f16) + bias + resid
// 256 threads = 8 waves; wave w -> rows [m0,m0+16), cols [n0,n0+64).
// B panel (64 x 32 per k-step) staged in LDS via async-to-LDS, double-buffered
// so the next panel streams in while WMMAs consume the current one.
// Grid: (N/64, M/128).
// ---------------------------------------------------------------------------
#define BSTRIDE 40   // padded LDS row stride (halfwords): conflict-free b128
__global__ __launch_bounds__(256, 2)
void gemm_wmma_kernel(const _Float16* __restrict__ A,
                      const _Float16* __restrict__ W,
                      const float* __restrict__ bias,
                      const float* __restrict__ resid,
                      _Float16* __restrict__ outh,
                      float* __restrict__ outf,
                      int M, int N, int K, int do_relu) {
    __shared__ __align__(64) _Float16 Bs[2][64][BSTRIDE];

    const int tid  = threadIdx.x;
    const int lane = tid & 31;
    const int wave = tid >> 5;
    const int half = lane >> 4;
    const int l16  = lane & 15;
    const int m0 = blockIdx.y * 128 + wave * 16;
    const int n0 = blockIdx.x * 64;

    // Cooperative staging: thread t copies 16B of W row (n0 + t/4), chunk t%4
    const int srow   = tid >> 2;
    const int schunk = (tid & 3) * 8;
    const _Float16* Wrow = W + (size_t)(n0 + srow) * K + schunk;
    const unsigned ldsa0 = lds_addr(&Bs[0][srow][schunk]);
    const unsigned ldsa1 = lds_addr(&Bs[1][srow][schunk]);

    v8f acc[4];
#pragma unroll
    for (int j = 0; j < 4; ++j)
#pragma unroll
        for (int r = 0; r < 8; ++r) acc[j][r] = 0.0f;

    const _Float16* Arow = A + (size_t)(m0 + l16) * K;

    // Prime buffer 0 with k-step 0
    async_copy16(ldsa0, Wrow);

    for (int kb = 0; kb < K; kb += 32) {
        const int buf = (kb >> 5) & 1;
        wait_async0();        // my 16B slice of current panel has landed
        __syncthreads();      // everyone's slice has landed; prev reads done
        if (kb + 32 < K)      // stream next panel into the other buffer
            async_copy16(buf ? ldsa0 : ldsa1, Wrow + kb + 32);
        if (kb + 32 < K)
            __builtin_prefetch(Arow + kb + 32, 0, 3);

        // A fragment (ISA 16-bit 16x32 layout): two contiguous 16B chunks
        v8h alo = *(const v8h*)(Arow + kb + 8 * half);
        v8h ahi = *(const v8h*)(Arow + kb + 16 + 8 * half);
        v16h a;
#pragma unroll
        for (int i = 0; i < 8; ++i) { a[i] = alo[i]; a[i + 8] = ahi[i]; }

#pragma unroll
        for (int j = 0; j < 4; ++j) {
            // B fragment from LDS: lane -> column n, K = 8*half+i and 16+8*half+i
            const _Float16* bp = &Bs[buf][j * 16 + l16][8 * half];
            v8h blo = *(const v8h*)bp;
            v8h bhi = *(const v8h*)(bp + 16);
            v16h b;
#pragma unroll
            for (int i = 0; i < 8; ++i) { b[i] = blo[i]; b[i + 8] = bhi[i]; }
            acc[j] = wmma_f16(a, b, acc[j]);
        }
    }

    // Epilogue: C layout -> lane(l16) = column, reg r -> row r + 8*half
#pragma unroll
    for (int j = 0; j < 4; ++j) {
        int n = n0 + j * 16 + l16;
        float bv = bias ? bias[n] : 0.0f;
#pragma unroll
        for (int r = 0; r < 8; ++r) {
            int m = m0 + r + 8 * half;
            float v = acc[j][r] + bv;
            if (resid) v += resid[(size_t)m * N + n];
            if (do_relu) v = v > 0.0f ? v : 0.0f;
            size_t o = (size_t)m * N + n;
            if (outf) outf[o] = v;
            if (outh) outh[o] = (_Float16)v;
        }
    }
}

// ---------------------------------------------------------------------------
// Flash-style causal attention, one wave per (batch, head, 16-query tile).
// qkv: (MROWS, 768) f16 rows = [q(256) | k(256) | v(256)], head h at h*32.
// V chunk is async-copied to LDS and the copy overlaps Q.K^T + softmax.
// out: (MROWS, 256) f16.
// ---------------------------------------------------------------------------
__global__ __launch_bounds__(128, 2)
void attn_kernel(const _Float16* __restrict__ qkv, _Float16* __restrict__ out) {
    __shared__ __align__(64) _Float16 Plds[4][16][32];
    __shared__ __align__(64) _Float16 Vlds[4][32][32];

    const int lane = threadIdx.x & 31;
    const int w    = threadIdx.x >> 5;
    const int half = lane >> 4;
    const int l16  = lane & 15;
    const int wid  = blockIdx.x * 4 + w;
    const int qt = wid & 127;
    const int h  = (wid >> 7) & 7;
    const int bb = wid >> 10;
    const int q0 = qt * 16;
    const size_t RS = 3 * DMODEL; // 768 row stride

    const unsigned vdst = lds_addr(&Vlds[w][lane][0]);  // 64B row per lane

    // Q fragment, pre-scaled by 1/sqrt(DH)
    v16h qa;
    {
        const _Float16* Qp =
            qkv + ((size_t)bb * SEQL + q0 + l16) * RS + h * DHEAD;
        v8h lo = *(const v8h*)(Qp + 16 * half);
        v8h hi = *(const v8h*)(Qp + 16 * half + 8);
        const _Float16 sc = (_Float16)0.17677669529663687f;
#pragma unroll
        for (int i = 0; i < 8; ++i) { qa[i] = lo[i] * sc; qa[i + 8] = hi[i] * sc; }
    }

    float mrow[8], lrow[8];
    v8f O0, O1;
#pragma unroll
    for (int r = 0; r < 8; ++r) {
        mrow[r] = -1e30f; lrow[r] = 0.0f; O0[r] = 0.0f; O1[r] = 0.0f;
    }

    for (int k0 = 0; k0 <= q0 + 15; k0 += 32) {
        // Kick off async V-chunk copy (32 keys x 32 dh); one key row per lane.
        {
            const _Float16* Vp = qkv +
                ((size_t)bb * SEQL + k0 + lane) * RS + 2 * DMODEL + h * DHEAD;
            async_copy16(vdst +  0, Vp + 0);
            async_copy16(vdst + 16, Vp + 8);
            async_copy16(vdst + 32, Vp + 16);
            async_copy16(vdst + 48, Vp + 24);
        }

        // K^T fragments: lane -> key column, 16 contiguous dh halfwords
        const _Float16* Kp0 =
            qkv + ((size_t)bb * SEQL + k0 + l16) * RS + DMODEL + h * DHEAD + 16 * half;
        v16h kb0 = *(const v16h*)Kp0;
        v16h kb1 = *(const v16h*)(Kp0 + 16 * RS);

        v8f s0, s1;
#pragma unroll
        for (int r = 0; r < 8; ++r) { s0[r] = 0.0f; s1[r] = 0.0f; }
        s0 = wmma_f16(qa, kb0, s0);
        s1 = wmma_f16(qa, kb1, s1);

        // Causal mask (only near the diagonal)
        if (k0 + 31 > q0) {
#pragma unroll
            for (int r = 0; r < 8; ++r) {
                int q = q0 + r + 8 * half;
                if (k0 + l16 > q)      s0[r] = -1e30f;
                if (k0 + 16 + l16 > q) s1[r] = -1e30f;
            }
        }

        // Online softmax: rows live across 16 lanes of each half-wave
#pragma unroll
        for (int r = 0; r < 8; ++r) {
            float rmax = fmaxf(s0[r], s1[r]);
            rmax = fmaxf(rmax, __shfl_xor(rmax, 1, 32));
            rmax = fmaxf(rmax, __shfl_xor(rmax, 2, 32));
            rmax = fmaxf(rmax, __shfl_xor(rmax, 4, 32));
            rmax = fmaxf(rmax, __shfl_xor(rmax, 8, 32));
            float mnew  = fmaxf(mrow[r], rmax);
            float alpha = __expf(mrow[r] - mnew);
            float p0 = __expf(s0[r] - mnew);
            float p1 = __expf(s1[r] - mnew);
            float ps = p0 + p1;
            ps += __shfl_xor(ps, 1, 32);
            ps += __shfl_xor(ps, 2, 32);
            ps += __shfl_xor(ps, 4, 32);
            ps += __shfl_xor(ps, 8, 32);
            lrow[r] = lrow[r] * alpha + ps;
            mrow[r] = mnew;
            O0[r] *= alpha;
            O1[r] *= alpha;
            int m = r + 8 * half;
            Plds[w][m][l16]      = (_Float16)p0;
            Plds[w][m][16 + l16] = (_Float16)p1;
        }
        asm volatile("" ::: "memory"); // keep LDS store->load program order

        // P as A fragment (16x32)
        v16h pa;
#pragma unroll
        for (int i = 0; i < 8; ++i) {
            pa[i]     = Plds[w][l16][16 * half + i];
            pa[i + 8] = Plds[w][l16][16 * half + 8 + i];
        }

        // V fragments: async copy has had S-wmma + softmax time to land
        wait_async0();
        v16h vb0, vb1;
#pragma unroll
        for (int i = 0; i < 16; ++i) {
            vb0[i] = Vlds[w][16 * half + i][l16];
            vb1[i] = Vlds[w][16 * half + i][16 + l16];
        }
        O0 = wmma_f16(pa, vb0, O0);
        O1 = wmma_f16(pa, vb1, O1);
    }

    _Float16* Op = out + ((size_t)bb * SEQL + q0) * DMODEL + h * DHEAD;
#pragma unroll
    for (int r = 0; r < 8; ++r) {
        int m = r + 8 * half;
        float inv = 1.0f / lrow[r];
        Op[(size_t)m * DMODEL + l16]      = (_Float16)(O0[r] * inv);
        Op[(size_t)m * DMODEL + 16 + l16] = (_Float16)(O1[r] * inv);
    }
}

// ---------------------------------------------------------------------------
// LayerNorm over D=256, one wave per row (8 rows per 256-thread block)
// ---------------------------------------------------------------------------
__global__ __launch_bounds__(256)
void ln_kernel(const float* __restrict__ in, const float* __restrict__ g,
               const float* __restrict__ bta, float* __restrict__ xf,
               _Float16* __restrict__ xh) {
    int lane = threadIdx.x & 31, w = threadIdx.x >> 5;
    int row = blockIdx.x * 8 + w;
    const float* p = in + (size_t)row * DMODEL;
    float vals[8], s = 0.0f, s2 = 0.0f;
#pragma unroll
    for (int i = 0; i < 8; ++i) {
        float v = p[lane + 32 * i];
        vals[i] = v; s += v; s2 += v * v;
    }
#pragma unroll
    for (int o = 1; o < 32; o <<= 1) {
        s  += __shfl_xor(s,  o, 32);
        s2 += __shfl_xor(s2, o, 32);
    }
    float mean = s * (1.0f / 256.0f);
    float var  = s2 * (1.0f / 256.0f) - mean * mean;
    float rstd = rsqrtf(var + 1e-5f);
#pragma unroll
    for (int i = 0; i < 8; ++i) {
        int d = lane + 32 * i;
        float v = (vals[i] - mean) * rstd * g[d] + bta[d];
        xf[(size_t)row * DMODEL + d] = v;
        xh[(size_t)row * DMODEL + d] = (_Float16)v;
    }
}

// ---------------------------------------------------------------------------
// Head: decoder dot, IDM scan (incl. reference's v0[-1] quirk), fusion
// ---------------------------------------------------------------------------
__global__ __launch_bounds__(256)
void head_kernel(const float* __restrict__ xf, const float* __restrict__ inputs,
                 const float* __restrict__ his, const float* __restrict__ W_dec,
                 const float* __restrict__ b_dec, const float* __restrict__ W_fus,
                 const float* __restrict__ b_fus, float* __restrict__ outp) {
    __shared__ float dec[NBATCH];
    __shared__ float yidm[NBATCH][LOUT];
    int lane = threadIdx.x & 31, w = threadIdx.x >> 5;

    // dec[b] = dot(enc[b,-1,:], W_dec) + b_dec   (wave w handles batch w)
    {
        const float* p = xf + ((size_t)w * SEQL + SEQL - 1) * DMODEL;
        float s = 0.0f;
#pragma unroll
        for (int i = 0; i < 8; ++i) s += p[lane + 32 * i] * W_dec[lane + 32 * i];
#pragma unroll
        for (int o = 1; o < 32; o <<= 1) s += __shfl_xor(s, o, 32);
        if (lane == 0) dec[w] = s + b_dec[0];
    }
    __syncthreads();

    // IDM scan: lanes 0..7 of wave 0, one batch per lane
    if (threadIdx.x < NBATCH) {
        int b = threadIdx.x;
        const float* il = inputs + ((size_t)b * SEQL + SEQL - 1) * 4;
        float y = il[0], v = il[1], sgap = il[2], dv = il[3];
        const float sqab2 = 2.0f * sqrtf(1.0f * 1.5f);  // 2*sqrt(A*B)
        float sx = 2.0f + fmaxf(0.0f, v * 1.5f + v * dv / sqab2);
        float t = v * (1.0f / 30.0f); t = t * t; t = t * t;
        float r2 = sx / sgap;
        float af = 1.0f * (1.0f - t - r2 * r2);
        float v0 = fmaxf(v + af * 0.1f, 0.0f);
        float v0last = __shfl(v0, 7, 32);   // reference uses v0[-1] for ALL batches
        float y0 = y + v0last * 0.1f;
        yidm[b][0] = y0;
        float vp = v0, yp = y0;
        for (int j = 0; j < LOUT - 1; ++j) {
            float hp = his[((size_t)b * LOUT + j) * 2 + 0];
            float hv = his[((size_t)b * LOUT + j) * 2 + 1];
            float dvj = hv - vp;
            float sj  = hp - yp;
            float sxj = 2.0f + fmaxf(0.0f, vp * 1.5f + vp * dvj / sqab2);
            float u = vp * (1.0f / 30.0f); u = u * u; u = u * u;
            float rr = sxj / sj;
            float acc = 1.0f * (1.0f - u - rr * rr);
            float v2 = vp + acc * 0.1f;
            v2 = (v2 <= 0.0f) ? 0.0f : v2;
            float y2 = yp + v2 * 0.1f;
            yidm[b][j + 1] = y2;
            vp = v2; yp = y2;
        }
    }
    __syncthreads();

    // Fusion
    for (int i = threadIdx.x; i < NBATCH * LOUT; i += 256) {
        int b = i / LOUT, j = i % LOUT;
        float lastpos = inputs[((size_t)b * SEQL + SEQL - 1) * 4];
        float prev    = inputs[((size_t)b * SEQL + SEQL - 1 - LOUT) * 4];
        float dvl = (lastpos - prev) * (1.0f / (float)LOUT);
        float hist = lastpos + dvl * (float)(j + 1);
        outp[i] = W_fus[0] * dec[b] + W_fus[1] * hist + W_fus[2] * yidm[b][j] +
                  b_fus[0];
    }
}

// ---------------------------------------------------------------------------
// Host orchestration
// ---------------------------------------------------------------------------
extern "C" void kernel_launch(void* const* d_in, const int* in_sizes, int n_in,
                              void* d_out, int out_size, void* d_ws,
                              size_t ws_size, hipStream_t stream) {
    (void)in_sizes; (void)n_in; (void)out_size; (void)ws_size;
    const float* inputs = (const float*)d_in[0];
    const float* his    = (const float*)d_in[1];
    const float* W_emb  = (const float*)d_in[2];
    const float* b_emb  = (const float*)d_in[3];
    const float* Wqkv   = (const float*)d_in[4];
    const float* bqkv   = (const float*)d_in[5];
    const float* Wo     = (const float*)d_in[6];
    const float* bo     = (const float*)d_in[7];
    const float* W1     = (const float*)d_in[8];
    const float* b1     = (const float*)d_in[9];
    const float* W2     = (const float*)d_in[10];
    const float* b2     = (const float*)d_in[11];
    const float* ln1_g  = (const float*)d_in[12];
    const float* ln1_b  = (const float*)d_in[13];
    const float* ln2_g  = (const float*)d_in[14];
    const float* ln2_b  = (const float*)d_in[15];
    const float* W_dec  = (const float*)d_in[16];
    const float* b_dec  = (const float*)d_in[17];
    const float* W_fus  = (const float*)d_in[18];
    const float* b_fus  = (const float*)d_in[19];
    float* outp = (float*)d_out;

    // Workspace layout (~86 MB; all sizes 256B-aligned)
    char* ws = (char*)d_ws;
    size_t off = 0;
    auto alloc = [&](size_t bytes) {
        void* p = ws + off;
        off = (off + bytes + 255) & ~(size_t)255;
        return p;
    };
    const size_t nWqkv = (size_t)NLAYERS * 3 * DMODEL * DMODEL;   // 786432
    const size_t nWo   = (size_t)NLAYERS * DMODEL * DMODEL;       // 262144
    const size_t nW1   = (size_t)NLAYERS * NHID * DMODEL;         // 1048576
    const size_t nW2   = (size_t)NLAYERS * DMODEL * NHID;         // 1048576

    _Float16* Wqkv16 = (_Float16*)alloc(nWqkv * 2);
    _Float16* Wo16   = (_Float16*)alloc(nWo * 2);
    _Float16* W116   = (_Float16*)alloc(nW1 * 2);
    _Float16* W216   = (_Float16*)alloc(nW2 * 2);
    float*    xf     = (float*)   alloc((size_t)MROWS * DMODEL * 4);
    _Float16* xh     = (_Float16*)alloc((size_t)MROWS * DMODEL * 2);
    _Float16* big    = (_Float16*)alloc((size_t)MROWS * NHID * 2); // qkv / ffn-h
    _Float16* qkv16  = big;
    _Float16* h16    = big;
    _Float16* ao16   = (_Float16*)alloc((size_t)MROWS * DMODEL * 2);
    float*    tmpf   = (float*)   alloc((size_t)MROWS * DMODEL * 4);

    // Convert weights to f16
    cvt_f16_kernel<<<(int)((nWqkv + 255) / 256), 256, 0, stream>>>(Wqkv, Wqkv16, (int)nWqkv);
    cvt_f16_kernel<<<(int)((nWo   + 255) / 256), 256, 0, stream>>>(Wo,   Wo16,   (int)nWo);
    cvt_f16_kernel<<<(int)((nW1   + 255) / 256), 256, 0, stream>>>(W1,   W116,   (int)nW1);
    cvt_f16_kernel<<<(int)((nW2   + 255) / 256), 256, 0, stream>>>(W2,   W216,   (int)nW2);

    // Embedding
    embed_kernel<<<MROWS, 256, 0, stream>>>(inputs, W_emb, b_emb, xf, xh);

    for (int l = 0; l < NLAYERS; ++l) {
        // QKV projection -> f16
        gemm_wmma_kernel<<<dim3(768 / 64, MROWS / 128), 256, 0, stream>>>(
            xh, Wqkv16 + (size_t)l * 3 * DMODEL * DMODEL, bqkv + l * 3 * DMODEL,
            nullptr, qkv16, nullptr, MROWS, 3 * DMODEL, DMODEL, 0);
        // Causal attention
        attn_kernel<<<(NBATCH * NHEADS * (SEQL / 16)) / 4, 128, 0, stream>>>(qkv16, ao16);
        // Output projection + residual -> tmpf, then LN1 -> xf/xh
        gemm_wmma_kernel<<<dim3(DMODEL / 64, MROWS / 128), 256, 0, stream>>>(
            ao16, Wo16 + (size_t)l * DMODEL * DMODEL, bo + l * DMODEL, xf,
            nullptr, tmpf, MROWS, DMODEL, DMODEL, 0);
        ln_kernel<<<MROWS / 8, 256, 0, stream>>>(tmpf, ln1_g + l * DMODEL,
                                                 ln1_b + l * DMODEL, xf, xh);
        // FFN1 (+ReLU) -> h16
        gemm_wmma_kernel<<<dim3(NHID / 64, MROWS / 128), 256, 0, stream>>>(
            xh, W116 + (size_t)l * NHID * DMODEL, b1 + l * NHID, nullptr, h16,
            nullptr, MROWS, NHID, DMODEL, 1);
        // FFN2 + residual -> tmpf, then LN2 -> xf/xh
        gemm_wmma_kernel<<<dim3(DMODEL / 64, MROWS / 128), 256, 0, stream>>>(
            h16, W216 + (size_t)l * DMODEL * NHID, b2 + l * DMODEL, xf, nullptr,
            tmpf, MROWS, DMODEL, NHID, 0);
        ln_kernel<<<MROWS / 8, 256, 0, stream>>>(tmpf, ln2_g + l * DMODEL,
                                                 ln2_b + l * DMODEL, xf, xh);
    }

    // Decoder + IDM + fusion
    head_kernel<<<1, 256, 0, stream>>>(xf, inputs, his, W_dec, b_dec, W_fus,
                                       b_fus, outp);
}
